// MultiHeadAttention_88785563943250
// MI455X (gfx1250) — matrix operand order repaired and checked by reading, not compiled, not run on hardware
//
#include <hip/hip_runtime.h>
#include <hip/hip_bf16.h>

typedef __attribute__((ext_vector_type(16))) __bf16 v16bf;
typedef __attribute__((ext_vector_type(8)))  float  v8f;

#define EMB   1024
#define HEADS 16
#define DHEAD 64
#define BATCH 4
#define SEQ   2048
#define ROWS  (BATCH * SEQ)   // 8192

#define LSX 40   // LDS row stride in bf16 elems (32 data + 8 pad -> 80B, 16B aligned)
#define PS  40   // P-tile row stride in bf16 elems

// ---- helpers -------------------------------------------------------------

__device__ __forceinline__ v16bf ldfrag(const __bf16* p0, const __bf16* p1) {
  union { uint4 u[2]; v16bf v; } r;
  r.u[0] = *(const uint4*)p0;
  r.u[1] = *(const uint4*)p1;
  return r.v;
}

__device__ __forceinline__ unsigned pk2(float a, float b) {
  union { __bf16 h[2]; unsigned u; } r;
  r.h[0] = (__bf16)a; r.h[1] = (__bf16)b;
  return r.u;
}

__device__ __forceinline__ v8f wmma_bf16(v16bf a, v16bf b, v8f c) {
  return __builtin_amdgcn_wmma_f32_16x16x32_bf16(false, a, false, b, (short)0, c,
                                                 false, false);
}

// Async global->LDS copy: each lane moves 16B.  GV mode (64-bit vaddr).
// LDS address operand = low 32 bits of the generic shared pointer (aperture
// rules).  Tracked by ASYNCcnt.
__device__ __forceinline__ void async_b128(const __bf16* g, __bf16* l) {
  asm volatile("global_load_async_to_lds_b128 %0, %1, off"
               :: "v"((unsigned)(size_t)l), "v"(g) : "memory");
}

// ---- Kernel 0: f32 -> bf16 bulk convert (8 elems / thread) ---------------
__global__ __launch_bounds__(256)
void f32_to_bf16(const float* __restrict__ src, __bf16* __restrict__ dst, int n)
{
  const int i = (blockIdx.x * 256 + threadIdx.x) * 8;
  if (i >= n) return;
  float4 a = *(const float4*)(src + i);
  float4 b = *(const float4*)(src + i + 4);
  uint4 p = { pk2(a.x,a.y), pk2(a.z,a.w), pk2(b.x,b.y), pk2(b.z,b.w) };
  *(uint4*)(dst + i) = p;
}

// ---- Kernel 1: projection GEMM  Y = X * W^T + bias  (bf16 in, bf16 out) --
// Xb: [ROWS, EMB] bf16 row-major.  Wb: [1024, EMB] bf16 row-major.
// mode 0: Y laid out [B*H, S, 64]   (Q, K)
// mode 1: Y laid out [B*H, 64, S]   (V transposed)
// Tiles stream global->LDS via async b128 copies, double-buffered.
__global__ __launch_bounds__(256)
void qkv_gemm_async(const __bf16* __restrict__ Xb, const __bf16* __restrict__ Wb,
                    const float* __restrict__ bias, __bf16* __restrict__ Y, int mode)
{
  __shared__ __bf16 Xs[2][64 * LSX];
  __shared__ __bf16 Ws[2][64 * LSX];

  const int tid  = threadIdx.x;
  const int wave = tid >> 5;
  const int lane = tid & 31;
  const int half = lane >> 4;
  const int lm   = lane & 15;

  const int rb = (blockIdx.x >> 4) * 64;   // 128 row blocks
  const int cb = (blockIdx.x & 15) * 64;   // 16 col blocks

  const int wrow = (wave >> 1) * 16;       // 0,16,32,48
  const int wcol = (wave & 1) * 32;        // 0,32

  // copy mapping: 256 threads x 16B cover one 64x32 bf16 tile
  const int lrow = tid >> 2;               // 0..63
  const int lseg = (tid & 3) * 8;          // elem offset 0,8,16,24

  const __bf16* xg = Xb + (size_t)(rb + lrow) * EMB + lseg;
  const __bf16* wg = Wb + (size_t)(cb + lrow) * EMB + lseg;
  __bf16* xl0 = &Xs[0][lrow * LSX + lseg];
  __bf16* wl0 = &Ws[0][lrow * LSX + lseg];
  __bf16* xl1 = &Xs[1][lrow * LSX + lseg];
  __bf16* wl1 = &Ws[1][lrow * LSX + lseg];

  v8f acc0 = {}; v8f acc1 = {};

  // prologue: fill buffer 0
  async_b128(xg, xl0);
  async_b128(wg, wl0);

  for (int k0 = 0; k0 < EMB; k0 += 32) {
    const int buf = (k0 >> 5) & 1;
    if (k0 + 32 < EMB) {
      // issue next tile into the other buffer, then retire current tile's copies
      async_b128(xg + (k0 + 32), buf ? xl0 : xl1);
      async_b128(wg + (k0 + 32), buf ? wl0 : wl1);
      asm volatile("s_wait_asynccnt 0x2" ::: "memory");
    } else {
      asm volatile("s_wait_asynccnt 0x0" ::: "memory");
    }
    __syncthreads();

    const __bf16* Xc = &Xs[buf][0];
    const __bf16* Wc = &Ws[buf][0];
    // A fragment (16x32): lane holds row lm, contiguous K = 16*half .. +15
    v16bf af  = ldfrag(&Xc[(wrow + lm) * LSX + 16*half],
                       &Xc[(wrow + lm) * LSX + 16*half + 8]);
    // B fragments (32x16): lane holds column lm, contiguous K = 16*half .. +15
    v16bf bf0 = ldfrag(&Wc[(wcol + lm) * LSX + 16*half],
                       &Wc[(wcol + lm) * LSX + 16*half + 8]);
    v16bf bf1 = ldfrag(&Wc[(wcol + 16 + lm) * LSX + 16*half],
                       &Wc[(wcol + 16 + lm) * LSX + 16*half + 8]);
    acc0 = wmma_bf16(af, bf0, acc0);
    acc1 = wmma_bf16(af, bf1, acc1);
    __syncthreads();   // all reads done before this buffer is refilled
  }

  // C/D layout: lane L, VGPR i -> row i + 8*(L/16), col L%16
  #pragma unroll
  for (int t = 0; t < 2; ++t) {
    v8f a = t ? acc1 : acc0;
    const int c = cb + wcol + t*16 + lm;
    const float bv = bias[c];
    const int h = c >> 6, d = c & 63;
    #pragma unroll
    for (int i = 0; i < 8; ++i) {
      const int r = rb + wrow + i + 8*half;
      const int b = r >> 11;            // r / SEQ
      const int s = r & (SEQ - 1);      // r % SEQ
      const float y = a[i] + bv;
      size_t idx;
      if (mode == 0)
        idx = ((size_t)(b * HEADS + h) * SEQ + s) * DHEAD + d;
      else
        idx = ((size_t)(b * HEADS + h) * DHEAD + d) * SEQ + s;
      Y[idx] = (__bf16)y;
    }
  }
}

// ---- Kernel 2: causal flash attention -----------------------------------
// Q,K: bf16 [B*H, S, 64].  VT: bf16 [B*H, 64, S].  out: f32 [B, S, H*64].
__global__ __launch_bounds__(128)
void flash_attn(const __bf16* __restrict__ Q, const __bf16* __restrict__ K,
                const __bf16* __restrict__ VT, float* __restrict__ out)
{
  __shared__ __bf16 Ps[4][16 * PS];   // wave-private P tiles (16 q x 32 keys)

  const int lane = threadIdx.x & 31;
  const int wave = threadIdx.x >> 5;
  const int half = lane >> 4;
  const int lm   = lane & 15;

  const int QBLK  = SEQ / 64;                    // 32
  const int bh    = blockIdx.x / QBLK;
  const int qb    = (blockIdx.x % QBLK) * 64;
  const int qbase = qb + wave * 16;

  const __bf16* qp = Q  + (size_t)bh * SEQ * DHEAD;
  const __bf16* kp = K  + (size_t)bh * SEQ * DHEAD;
  const __bf16* vp = VT + (size_t)bh * DHEAD * SEQ;

  // Q A-fragments for d-chunks 0..31 and 32..63 (held for whole key loop)
  const __bf16* qrow = qp + (size_t)(qbase + lm) * DHEAD;
  v16bf qa0 = ldfrag(qrow +      16*half, qrow +  8 + 16*half);
  v16bf qa1 = ldfrag(qrow + 32 + 16*half, qrow + 40 + 16*half);

  float m_i[8], l_i[8], al[8];
  #pragma unroll
  for (int i = 0; i < 8; ++i) { m_i[i] = -1e30f; l_i[i] = 0.f; }
  v8f acc[4] = {};   // O accumulators, 4 vdim tiles of 16

  __bf16* Pw = &Ps[wave][0];

  const int nkb = (qbase + 16 + 31) >> 5;   // causal: key blocks of 32
  for (int kb = 0; kb < nkb; ++kb) {
    const int k0 = kb * 32;

    // K^T B-fragments: lane = key k0 + t*16 + lm, contiguous d = dc + 16*half .. +15
    const __bf16* kr0 = kp + (size_t)(k0 +      lm) * DHEAD + 16*half;
    const __bf16* kr1 = kp + (size_t)(k0 + 16 + lm) * DHEAD + 16*half;
    v16bf kb0d0 = ldfrag(kr0,      kr0 + 8);
    v16bf kb0d1 = ldfrag(kr0 + 32, kr0 + 40);
    v16bf kb1d0 = ldfrag(kr1,      kr1 + 8);
    v16bf kb1d1 = ldfrag(kr1 + 32, kr1 + 40);

    v8f c0 = {}; v8f c1 = {};
    c0 = wmma_bf16(qa0, kb0d0, c0);
    c0 = wmma_bf16(qa1, kb0d1, c0);
    c1 = wmma_bf16(qa0, kb1d0, c1);
    c1 = wmma_bf16(qa1, kb1d1, c1);

    if (kb + 1 < nkb)   // global_prefetch_b8 of next key block
      __builtin_prefetch(kp + (size_t)(k0 + 32 + lm) * DHEAD, 0, 1);

    // scale 1/sqrt(64), causal mask, block row-max
    float bm[8];
    #pragma unroll
    for (int i = 0; i < 8; ++i) {
      const int qr = qbase + i + 8*half;
      float s0 = c0[i] * 0.125f;
      float s1 = c1[i] * 0.125f;
      if (k0 +      lm > qr) s0 = -1e30f;
      if (k0 + 16 + lm > qr) s1 = -1e30f;
      c0[i] = s0; c1[i] = s1;
      bm[i] = fmaxf(s0, s1);
    }
    #pragma unroll
    for (int m = 1; m < 16; m <<= 1)
      #pragma unroll
      for (int i = 0; i < 8; ++i)
        bm[i] = fmaxf(bm[i], __shfl_xor(bm[i], m, 32));

    // online softmax update
    float rs[8];
    #pragma unroll
    for (int i = 0; i < 8; ++i) {
      const float mn = fmaxf(m_i[i], bm[i]);
      al[i] = __expf(m_i[i] - mn);
      m_i[i] = mn;
      c0[i] = __expf(c0[i] - mn);
      c1[i] = __expf(c1[i] - mn);
      rs[i] = c0[i] + c1[i];
    }
    #pragma unroll
    for (int m = 1; m < 16; m <<= 1)
      #pragma unroll
      for (int i = 0; i < 8; ++i)
        rs[i] += __shfl_xor(rs[i], m, 32);
    #pragma unroll
    for (int i = 0; i < 8; ++i) l_i[i] = l_i[i] * al[i] + rs[i];

    #pragma unroll
    for (int v = 0; v < 4; ++v)
      #pragma unroll
      for (int i = 0; i < 8; ++i)
        acc[v][i] *= al[i];

    // P (C-layout) -> LDS -> A-fragment layout for PV
    #pragma unroll
    for (int i = 0; i < 8; ++i) {
      const int r = i + 8*half;
      Pw[r * PS +      lm] = (__bf16)c0[i];
      Pw[r * PS + 16 + lm] = (__bf16)c1[i];
    }
    asm volatile("s_wait_dscnt 0x0" ::: "memory");   // same-wave LDS RAW
    v16bf pa = ldfrag(&Pw[lm * PS + 16*half], &Pw[lm * PS + 16*half + 8]);

    // PV: B-fragments from transposed V (contiguous 16 keys per half-wave)
    #pragma unroll
    for (int v = 0; v < 4; ++v) {
      const __bf16* vr = vp + (size_t)(v*16 + lm) * SEQ + k0 + 16*half;
      v16bf vb = ldfrag(vr, vr + 8);
      acc[v] = wmma_bf16(pa, vb, acc[v]);
    }
  }

  // epilogue: normalize and store f32 out[b, s, h*64 + vd]
  const int b = bh / HEADS, h = bh % HEADS;
  #pragma unroll
  for (int i = 0; i < 8; ++i) al[i] = 1.0f / l_i[i];
  #pragma unroll
  for (int v = 0; v < 4; ++v)
    #pragma unroll
    for (int i = 0; i < 8; ++i) {
      const int srow = qbase + i + 8*half;
      const int vd   = v*16 + lm;
      out[(((size_t)b * SEQ + srow) * HEADS + h) * DHEAD + vd] = acc[v][i] * al[i];
    }
}

// ---- launcher ------------------------------------------------------------

extern "C" void kernel_launch(void* const* d_in, const int* in_sizes, int n_in,
                              void* d_out, int out_size, void* d_ws, size_t ws_size,
                              hipStream_t stream)
{
  (void)in_sizes; (void)n_in; (void)out_size; (void)ws_size;
  const float* x_q  = (const float*)d_in[0];
  const float* x_kv = (const float*)d_in[1];
  // d_in[2] = attn_mask (causal) — applied analytically in flash_attn
  const float* w_q = (const float*)d_in[3];
  const float* b_q = (const float*)d_in[4];
  const float* w_k = (const float*)d_in[5];
  const float* b_k = (const float*)d_in[6];
  const float* w_v = (const float*)d_in[7];
  const float* b_v = (const float*)d_in[8];

  const size_t NX = (size_t)ROWS * EMB;     // 8M elems
  const size_t NW = (size_t)EMB * EMB;      // 1M elems

  __bf16* qws = (__bf16*)d_ws;              // [B*H, S, 64]
  __bf16* kws = qws + NX;                   // [B*H, S, 64]
  __bf16* vws = kws + NX;                   // [B*H, 64, S]
  __bf16* xqb = vws + NX;                   // x_q   bf16
  __bf16* xkb = xqb + NX;                   // x_k_v bf16
  __bf16* wqb = xkb + NX;                   // weights bf16
  __bf16* wkb = wqb + NW;
  __bf16* wvb = wkb + NW;

  // one-time f32 -> bf16 conversions
  f32_to_bf16<<<NX / 2048, 256, 0, stream>>>(x_q,  xqb, (int)NX);
  f32_to_bf16<<<NX / 2048, 256, 0, stream>>>(x_kv, xkb, (int)NX);
  f32_to_bf16<<<NW / 2048, 256, 0, stream>>>(w_q,  wqb, (int)NW);
  f32_to_bf16<<<NW / 2048, 256, 0, stream>>>(w_k,  wkb, (int)NW);
  f32_to_bf16<<<NW / 2048, 256, 0, stream>>>(w_v,  wvb, (int)NW);

  const dim3 gg((ROWS / 64) * (EMB / 64));  // 2048 workgroups
  qkv_gemm_async<<<gg, 256, 0, stream>>>(xqb, wqb, b_q, qws, 0);
  qkv_gemm_async<<<gg, 256, 0, stream>>>(xkb, wkb, b_k, kws, 0);
  qkv_gemm_async<<<gg, 256, 0, stream>>>(xkb, wvb, b_v, vws, 1);

  flash_attn<<<BATCH * HEADS * (SEQ / 64), 128, 0, stream>>>(
      qws, kws, vws, (float*)d_out);
}
